// ECPEC_74672301408834
// MI455X (gfx1250) — matrix operand order repaired and checked by reading, not compiled, not run on hardware
//
#include <hip/hip_runtime.h>
#include <hip/hip_bf16.h>
#include <math.h>

// ------------------------------------------------------------------
// ECPEC forward for MI455X (gfx1250, wave32).
// All heavy GEMMs run through v_wmma_f32_16x16x32_f16 (f16 in, f32
// acc).  B matrices are kept in [N][K] (out x in) layout -- the
// native layout of every weight in this model -- which makes both A
// and B WMMA fragments per-lane contiguous 16-byte chunks, fed
// directly from global memory (192MB L2 absorbs reuse). No LDS, no
// barriers in the GEMM.  Wave tile 32x64 = 8 wmma per K-step.
// ------------------------------------------------------------------

typedef _Float16 half_t;
typedef __attribute__((ext_vector_type(16))) _Float16 v16h;
typedef __attribute__((ext_vector_type(8)))  float    v8f;
typedef __attribute__((ext_vector_type(8)))  _Float16 h8;

constexpr int Dd  = 768;
constexpr int Hh  = 384;
constexpr int G4  = 1536;           // 4*H
constexpr int Tt  = 64;
constexpr int Bb  = 512;
constexpr int NE  = 64;
constexpr int NC  = 256;
constexpr int NCH = 16;             // N_CAU / CK
constexpr int PS  = 64;             // POS
constexpr int BT  = Bb * Tt;        // 32768
constexpr int KD2 = 3 * Dd + 1 + PS;  // 2369
constexpr int KD3 = 3 * Dd + 2 + PS;  // 2370
constexpr int KP  = 2400;           // padded K (multiple of 32)

#define FLAG_BIAS  1
#define FLAG_LEAKY 2

// ========================= WMMA GEMM ==============================
// C[M x N] (f32) = act( A[M x K](f16 rowmajor) * Bt[N x K](f16 rowmajor)^T + bias )
// Requirements: N % 64 == 0, K % 32 == 0, A/Bt 16-byte aligned.  M arbitrary
// (rows are clamp-loaded / guard-stored).
__global__ __launch_bounds__(256) void wmma_gemm(
    const half_t* __restrict__ A, const half_t* __restrict__ Bt,
    const float* __restrict__ bias, float* __restrict__ C,
    int M, int N, int K, int flags)
{
  const int wave = threadIdx.x >> 5;
  const int lane = threadIdx.x & 31;
  const int r    = lane & 15;          // row-in-16 (A) / col-in-16 (B/C)
  const int hi   = lane >> 4;          // half-wave select
  const int n0   = blockIdx.x * 64;
  const int mw   = blockIdx.y * 256 + wave * 32;

  int ar0 = mw + r;       if (ar0 >= M) ar0 = M - 1;
  int ar1 = mw + 16 + r;  if (ar1 >= M) ar1 = M - 1;
  // A fragment (16x32 f16): lane holds row, elems 0..7 at K=hi*8+.., 8..15 at K=16+hi*8+..
  const half_t* Ap0 = A + (size_t)ar0 * K + hi * 8;
  const half_t* Ap1 = A + (size_t)ar1 * K + hi * 8;
  // B fragment (32x16 f16): lane holds column n, elems 0..7 at K=hi*8+.., 8..15 at K=16+hi*8+..
  const half_t* Bp  = Bt + (size_t)(n0 + r) * K + hi * 8;

  const v8f vzero = {0.f, 0.f, 0.f, 0.f, 0.f, 0.f, 0.f, 0.f};
  v8f acc[2][4];
#pragma unroll
  for (int i = 0; i < 2; ++i)
#pragma unroll
    for (int f = 0; f < 4; ++f) acc[i][f] = vzero;

  union HF { v16h v; h8 h[2]; };

  for (int k0 = 0; k0 < K; k0 += 32) {
    HF a0, a1;
    a0.h[0] = *(const h8*)(Ap0 + k0);
    a0.h[1] = *(const h8*)(Ap0 + k0 + 16);
    a1.h[0] = *(const h8*)(Ap1 + k0);
    a1.h[1] = *(const h8*)(Ap1 + k0 + 16);
#pragma unroll
    for (int f = 0; f < 4; ++f) {
      HF b;
      const half_t* bp = Bp + (size_t)(f * 16) * K + k0;
      b.h[0] = *(const h8*)(bp);
      b.h[1] = *(const h8*)(bp + 16);
      acc[0][f] = __builtin_amdgcn_wmma_f32_16x16x32_f16(false, a0.v, false, b.v,
                                                         (short)0, acc[0][f], false, false);
      acc[1][f] = __builtin_amdgcn_wmma_f32_16x16x32_f16(false, a1.v, false, b.v,
                                                         (short)0, acc[1][f], false, false);
    }
    if (k0 + 32 < K) {
      __builtin_prefetch((const void*)(Ap0 + k0 + 32), 0, 1);
      __builtin_prefetch((const void*)(Ap1 + k0 + 32), 0, 1);
      __builtin_prefetch((const void*)(Bp + k0 + 32), 0, 1);
    }
  }

#pragma unroll
  for (int i = 0; i < 2; ++i)
#pragma unroll
    for (int f = 0; f < 4; ++f)
#pragma unroll
      for (int v = 0; v < 8; ++v) {
        int m = mw + i * 16 + hi * 8 + v;     // C: lane holds col r, vgpr v -> row v + hi*8
        int n = n0 + f * 16 + r;
        if (m < M) {
          float val = acc[i][f][v];
          if (flags & FLAG_BIAS)  val += bias[n];
          if (flags & FLAG_LEAKY) val = val > 0.f ? val : 0.01f * val;
          C[(size_t)m * N + n] = val;
        }
      }
}

// ======================= small helpers ============================
__global__ void k_f32_to_f16(const float* __restrict__ s, half_t* __restrict__ d, int n) {
  int i = blockIdx.x * blockDim.x + threadIdx.x;
  if (i < n) d[i] = (half_t)s[i];
}
// dst[n*Kp + k] = (k < Ksrc) ? f16(src[n*Ksrc + k]) : 0     (K-pad, no transpose)
__global__ void k_pad_f16(const float* __restrict__ src, half_t* __restrict__ dst,
                          int Ksrc, int Kp, int N) {
  int i = blockIdx.x * blockDim.x + threadIdx.x;
  if (i >= N * Kp) return;
  int n = i / Kp, k = i - n * Kp;
  dst[i] = (k < Ksrc) ? (half_t)src[(size_t)n * Ksrc + k] : (half_t)0.f;
}
__global__ void k_zero_f32(float* p, int n) {
  int i = blockIdx.x * blockDim.x + threadIdx.x;
  if (i < n) p[i] = 0.f;
}
__global__ void k_zero_f16(half_t* p, int n) {
  int i = blockIdx.x * blockDim.x + threadIdx.x;
  if (i < n) p[i] = (half_t)0.f;
}

__device__ __forceinline__ float sigf(float x) { return 1.f / (1.f + __expf(-x)); }

// ======================= LSTM pointwise ===========================
__global__ void k_lstm_step(const float* __restrict__ gf, const float* __restrict__ gr,
                            const float* __restrict__ Gf, const float* __restrict__ Gxr,
                            const int* __restrict__ len, int t,
                            float* __restrict__ cF, float* __restrict__ cR,
                            float* __restrict__ hFh, float* __restrict__ hRh,
                            half_t* __restrict__ h16F, half_t* __restrict__ h16R)
{
  int tid = blockIdx.x * blockDim.x + threadIdx.x;
  if (tid >= 2 * Bb * Hh) return;
  int dir = tid / (Bb * Hh);
  int rem = tid - dir * Bb * Hh;
  int b = rem / Hh;
  int j = rem - b * Hh;
  int gbase = b * G4 + j;
  float gi, gfo, gg, go;
  if (dir == 0) {
    size_t Gb = ((size_t)b * Tt + t) * G4;
    gi  = gf[gbase]          + Gf[Gb + j];
    gfo = gf[gbase + Hh]     + Gf[Gb + Hh + j];
    gg  = gf[gbase + 2 * Hh] + Gf[Gb + 2 * Hh + j];
    go  = gf[gbase + 3 * Hh] + Gf[Gb + 3 * Hh + j];
    float c = sigf(gfo) * cF[rem] + sigf(gi) * tanhf(gg);
    float h = sigf(go) * tanhf(c);
    cF[rem] = c;
    hFh[((size_t)b * Tt + t) * Hh + j] = h;
    h16F[rem] = (half_t)h;
  } else {
    int L = len[b];
    gi  = gr[gbase];
    gfo = gr[gbase + Hh];
    gg  = gr[gbase + 2 * Hh];
    go  = gr[gbase + 3 * Hh];
    if (t < L) {  // gathered, masked input gates (bias b_r comes via gr GEMM)
      size_t Gb = ((size_t)b * Tt + (L - 1 - t)) * G4;
      gi  += Gxr[Gb + j];
      gfo += Gxr[Gb + Hh + j];
      gg  += Gxr[Gb + 2 * Hh + j];
      go  += Gxr[Gb + 3 * Hh + j];
    }
    float c = sigf(gfo) * cR[rem] + sigf(gi) * tanhf(gg);
    float h = sigf(go) * tanhf(c);
    cR[rem] = c;
    hRh[((size_t)b * Tt + t) * Hh + j] = h;
    h16R[rem] = (half_t)h;
  }
}

// enc = concat(out_f, out_b) with masking and reverse un-gather
__global__ void k_build_enc(const float* __restrict__ hFh, const float* __restrict__ hRh,
                            const int* __restrict__ len,
                            float* __restrict__ enc, half_t* __restrict__ enc16)
{
  int tid = blockIdx.x * blockDim.x + threadIdx.x;
  if (tid >= BT * Dd) return;
  int b = tid / (Tt * Dd);
  int rem = tid - b * (Tt * Dd);
  int t = rem / Dd;
  int d = rem - t * Dd;
  int L = len[b];
  float v = 0.f;
  if (t < L) {
    if (d < Hh) v = hFh[((size_t)b * Tt + t) * Hh + d];
    else        v = hRh[((size_t)b * Tt + (L - 1 - t)) * Hh + (d - Hh)];
  }
  enc[tid] = v;
  enc16[tid] = (half_t)v;
}

// ================== attention (wave-per-row dots) =================
__global__ __launch_bounds__(256) void k_alpha_raw(const float* __restrict__ Z,
                                                   const float* __restrict__ s2,
                                                   const int* __restrict__ len,
                                                   float* __restrict__ araw)
{
  int wave = threadIdx.x >> 5, lane = threadIdx.x & 31;
  int row = blockIdx.x * 8 + wave;          // row < BT
  const float* zr = Z + (size_t)row * Dd;
  float s = 0.f;
  for (int i = lane; i < Dd; i += 32) s += zr[i] * s2[i];
#pragma unroll
  for (int off = 16; off; off >>= 1) s += __shfl_xor(s, off, 32);
  if (lane == 0) {
    int b = row >> 6, t = row & 63;
    araw[row] = (t < len[b]) ? s : 0.f;
  }
}

__global__ __launch_bounds__(64) void k_alpha_norm(const float* __restrict__ araw,
                                                   float* __restrict__ alpha)
{
  __shared__ float sh[64];
  int t = threadIdx.x, b = blockIdx.x;
  float v = araw[b * Tt + t];
  sh[t] = v;
  __syncthreads();
#pragma unroll
  for (int off = 32; off; off >>= 1) {
    if (t < off) sh[t] += sh[t + off];
    __syncthreads();
  }
  alpha[b * Tt + t] = v / (sh[0] + 1e-9f);
}

__global__ void k_weighted_sum(const float* __restrict__ alpha, const float* __restrict__ enc,
                               float* __restrict__ U)
{
  int tid = blockIdx.x * blockDim.x + threadIdx.x;
  if (tid >= Bb * Dd) return;
  int b = tid / Dd, d = tid - b * Dd;
  float s = 0.f;
  for (int t = 0; t < Tt; ++t)
    s += alpha[b * Tt + t] * enc[((size_t)b * Tt + t) * Dd + d];
  U[tid] = s;
}

// ordered nonzero indices
__global__ void k_indices(const int* __restrict__ be, const int* __restrict__ bc,
                          int* __restrict__ eidx, int* __restrict__ cidx)
{
  if (threadIdx.x != 0 || blockIdx.x != 0) return;
  int e = 0, c = 0;
  for (int b = 0; b < Bb; ++b) {
    if (be[b] && e < NE) eidx[e++] = b;
    if (bc[b] && c < NC) cidx[c++] = b;
  }
  while (e < NE) eidx[e++] = 0;
  while (c < NC) cidx[c++] = 0;
}

__global__ void k_gather_U(const float* __restrict__ U, const int* __restrict__ eidx,
                           const int* __restrict__ cidx,
                           float* __restrict__ Ug, half_t* __restrict__ Ug16)
{
  int tid = blockIdx.x * blockDim.x + threadIdx.x;
  if (tid >= (NE + NC) * Dd) return;
  int r = tid / Dd, d = tid - r * Dd;
  int src = (r < NE) ? eidx[r] : cidx[r - NE];
  float v = U[(size_t)src * Dd + d];
  Ug[tid] = v;
  Ug16[tid] = (half_t)v;
}

// text = share ? leaky(U@rep^T+b) (already in reptmp) : U;  f16 copy for GEMMs
__global__ void k_select_text(const float* __restrict__ reptmp, const float* __restrict__ Ug,
                              const int* __restrict__ share,
                              float* __restrict__ text, half_t* __restrict__ text16)
{
  int tid = blockIdx.x * blockDim.x + threadIdx.x;
  if (tid >= (NE + NC) * Dd) return;
  float v = (*share != 0) ? reptmp[tid] : Ug[tid];
  text[tid] = v;
  text16[tid] = (half_t)v;
}

// softmax over each chunk of 16 along k
__global__ void k_chunk_softmax(float* __restrict__ sc)
{
  int tid = blockIdx.x * blockDim.x + threadIdx.x;
  if (tid >= NE * NCH) return;
  int base = (tid / NCH) * NC + (tid % NCH) * 16;
  float m = -1e30f;
  for (int k = 0; k < 16; ++k) m = fmaxf(m, sc[base + k]);
  float s = 0.f;
  for (int k = 0; k < 16; ++k) { float e = __expf(sc[base + k] - m); sc[base + k] = e; s += e; }
  float inv = 1.f / s;
  for (int k = 0; k < 16; ++k) sc[base + k] *= inv;
}

__global__ void k_chunkemb(const float* __restrict__ sc, const float* __restrict__ text,
                           float* __restrict__ ce)
{
  int tid = blockIdx.x * blockDim.x + threadIdx.x;
  if (tid >= NE * NCH * Dd) return;
  int row = tid / Dd, d = tid - row * Dd;     // row = e*16 + n
  int e = row / NCH, n = row - e * NCH;
  float s = 0.f;
  for (int k = 0; k < 16; ++k)
    s += sc[e * NC + n * 16 + k] * text[(size_t)(NE + n * 16 + k) * Dd + d];
  ce[tid] = s;
}

// wave-per-row L2 norms
__global__ __launch_bounds__(256) void k_nrm2(const float* __restrict__ text,
                                              const float* __restrict__ ce,
                                              float* __restrict__ nrm)
{
  int wave = threadIdx.x >> 5, lane = threadIdx.x & 31;
  int row = blockIdx.x * 8 + wave;            // < 1024
  int e = row / NCH;
  float s = 0.f;
  for (int d = lane; d < Dd; d += 32) {
    float df = text[(size_t)e * Dd + d] - ce[(size_t)row * Dd + d];
    s += df * df;
  }
#pragma unroll
  for (int off = 16; off; off >>= 1) s += __shfl_xor(s, off, 32);
  if (lane == 0) nrm[row] = sqrtf(s);
}

__global__ __launch_bounds__(256) void k_nrm3(const float* __restrict__ text,
                                              float* __restrict__ nrm)
{
  int wave = threadIdx.x >> 5, lane = threadIdx.x & 31;
  int row = blockIdx.x * 8 + wave;            // < 16384, row = e*256 + c
  int e = row / NC, c = row - e * NC;
  float s = 0.f;
  for (int d = lane; d < Dd; d += 32) {
    float df = text[(size_t)e * Dd + d] - text[(size_t)(NE + c) * Dd + d];
    s += df * df;
  }
#pragma unroll
  for (int off = 16; off; off >>= 1) s += __shfl_xor(s, off, 32);
  if (lane == 0) nrm[row] = sqrtf(s);
}

// delta2: [xb | y | nrm | xb*y | pos] padded to KP, f16
__global__ void k_delta2(const float* __restrict__ text, const float* __restrict__ ce,
                         const float* __restrict__ nrm, const float* __restrict__ pos,
                         half_t* __restrict__ delta)
{
  int tid = blockIdx.x * blockDim.x + threadIdx.x;
  if (tid >= NE * NCH * KP) return;
  int row = tid / KP, col = tid - row * KP;   // row = e*16 + n
  int e = row / NCH, n = row - e * NCH;
  float v;
  if (col < Dd)            v = text[(size_t)e * Dd + col];
  else if (col < 2 * Dd)   v = ce[(size_t)row * Dd + (col - Dd)];
  else if (col == 2 * Dd)  v = nrm[row];
  else if (col < 3 * Dd + 1) {
    int d = col - (2 * Dd + 1);
    v = text[(size_t)e * Dd + d] * ce[(size_t)row * Dd + d];
  } else if (col < KD2)    v = pos[(size_t)(e * NCH + n) * PS + (col - (3 * Dd + 1))];
  else                     v = 0.f;
  delta[tid] = (half_t)v;
}

// phase2 head: 768->2, log_softmax, argmax side output
__global__ __launch_bounds__(256) void k_phase2(const float* __restrict__ Z2,
                                                const float* __restrict__ wo_w,
                                                const float* __restrict__ wo_b,
                                                float* __restrict__ out,
                                                float* __restrict__ extra)
{
  int wave = threadIdx.x >> 5, lane = threadIdx.x & 31;
  int row = blockIdx.x * 8 + wave;            // < 1024
  const float* z = Z2 + (size_t)row * Dd;
  float s0 = 0.f, s1 = 0.f;
  for (int i = lane; i < Dd; i += 32) { s0 += z[i] * wo_w[i]; s1 += z[i] * wo_w[Dd + i]; }
#pragma unroll
  for (int off = 16; off; off >>= 1) { s0 += __shfl_xor(s0, off, 32); s1 += __shfl_xor(s1, off, 32); }
  if (lane == 0) {
    float l0 = s0 + wo_b[0], l1 = s1 + wo_b[1];
    float m = fmaxf(l0, l1);
    float lse = m + __logf(__expf(l0 - m) + __expf(l1 - m));
    out[row * 2]     = l0 - lse;
    out[row * 2 + 1] = l1 - lse;
    extra[row] = (l1 > l0) ? 1.f : 0.f;
  }
}

// delta3: [xb | y | nrm | xb*y | dis | extra] padded to KP, f16
__global__ void k_delta3(const float* __restrict__ text, const float* __restrict__ nrm,
                         const float* __restrict__ dis, const float* __restrict__ extra,
                         half_t* __restrict__ delta)
{
  long long tid = (long long)blockIdx.x * blockDim.x + threadIdx.x;
  if (tid >= (long long)NE * NC * KP) return;
  int row = (int)(tid / KP), col = (int)(tid - (long long)row * KP);  // row = e*256 + c
  int e = row / NC, c = row - e * NC;
  float v;
  if (col < Dd)            v = text[(size_t)e * Dd + col];
  else if (col < 2 * Dd)   v = text[(size_t)(NE + c) * Dd + (col - Dd)];
  else if (col == 2 * Dd)  v = nrm[row];
  else if (col < 3 * Dd + 1) {
    int d = col - (2 * Dd + 1);
    v = text[(size_t)e * Dd + d] * text[(size_t)(NE + c) * Dd + d];
  } else if (col < 3 * Dd + 1 + PS)
    v = dis[((size_t)e * NC + c) * PS + (col - (3 * Dd + 1))];
  else if (col == KD3 - 1) v = extra[e * NCH + (c >> 4)];
  else                     v = 0.f;
  delta[tid] = (half_t)v;
}

// p head: 768->2, log_softmax
__global__ __launch_bounds__(256) void k_pcls(const float* __restrict__ Z3,
                                              const float* __restrict__ cls_w,
                                              const float* __restrict__ cls_b,
                                              float* __restrict__ out)
{
  int wave = threadIdx.x >> 5, lane = threadIdx.x & 31;
  int row = blockIdx.x * 8 + wave;            // < 16384
  const float* z = Z3 + (size_t)row * Dd;
  float s0 = 0.f, s1 = 0.f;
  for (int i = lane; i < Dd; i += 32) { s0 += z[i] * cls_w[i]; s1 += z[i] * cls_w[Dd + i]; }
#pragma unroll
  for (int off = 16; off; off >>= 1) { s0 += __shfl_xor(s0, off, 32); s1 += __shfl_xor(s1, off, 32); }
  if (lane == 0) {
    float l0 = s0 + cls_b[0], l1 = s1 + cls_b[1];
    float m = fmaxf(l0, l1);
    float lse = m + __logf(__expf(l0 - m) + __expf(l1 - m));
    out[row * 2]     = l0 - lse;
    out[row * 2 + 1] = l1 - lse;
  }
}

__global__ void k_labels(const int* __restrict__ label3, float* __restrict__ L)
{
  int e = blockIdx.x * blockDim.x + threadIdx.x;
  if (e >= NE) return;
  float* row = L + (size_t)e * NC;
  for (int c = 0; c < NC; ++c) row[c] = 0.f;
  for (int k = 0; k < 3; ++k) {
    int l = label3[e * 3 + k];
    if (l >= 0) row[l] = 1.f;   // .at[idx].max(1.0)
  }
}

// ============================ driver ==============================
#define CDIV(a, b) (((a) + (b) - 1) / (b))

static inline void gemm(hipStream_t s, const half_t* A, const half_t* Bt,
                        const float* bias, float* C, int M, int N, int K, int flags)
{
  dim3 g(N / 64, CDIV(M, 256));
  wmma_gemm<<<g, 256, 0, s>>>(A, Bt, bias, C, M, N, K, flags);
}

extern "C" void kernel_launch(void* const* d_in, const int* in_sizes, int n_in,
                              void* d_out, int out_size, void* d_ws, size_t ws_size,
                              hipStream_t stream)
{
  const float* word  = (const float*)d_in[0];
  const float* pos   = (const float*)d_in[1];
  const float* dis   = (const float*)d_in[2];
  const int*   slen  = (const int*)d_in[3];
  const int*   bemo  = (const int*)d_in[4];
  const int*   bcau  = (const int*)d_in[5];
  const int*   lab3  = (const int*)d_in[8];
  const int*   share = (const int*)d_in[9];
  const float* rep_w = (const float*)d_in[10];
  const float* rep_b = (const float*)d_in[11];
  const float* w2_w  = (const float*)d_in[12];
  const float* w2_b  = (const float*)d_in[13];
  const float* wo_w  = (const float*)d_in[14];
  const float* wo_b  = (const float*)d_in[15];
  const float* w3_w  = (const float*)d_in[16];
  const float* w3_b  = (const float*)d_in[17];
  const float* cls_w = (const float*)d_in[18];
  const float* cls_b = (const float*)d_in[19];
  const float* s1_w  = (const float*)d_in[20];
  const float* s1_b  = (const float*)d_in[21];
  const float* s2_w  = (const float*)d_in[22];
  const float* wihf  = (const float*)d_in[23];
  const float* whhf  = (const float*)d_in[24];
  const float* bf    = (const float*)d_in[25];
  const float* wihr  = (const float*)d_in[26];
  const float* whhr  = (const float*)d_in[27];
  const float* br    = (const float*)d_in[28];

  float* out = (float*)d_out;
  float* out_phase2 = out;                       // 64*16*2
  float* out_p      = out + NE * NCH * 2;        // 64*256*2
  float* out_L      = out_p + NE * NC * 2;       // 64*256

  char* base = (char*)d_ws;
  size_t off = 0;
  auto alloc = [&](size_t bytes) -> void* {
    void* p = base + off;
    off = (off + bytes + 255) & ~(size_t)255;
    return p;
  };

  half_t* Xf16    = (half_t*)alloc((size_t)BT * Dd * 2);
  half_t* wihf16  = (half_t*)alloc((size_t)G4 * Dd * 2);   // [N=1536][K=768]
  half_t* wihr16  = (half_t*)alloc((size_t)G4 * Dd * 2);
  half_t* whhf16  = (half_t*)alloc((size_t)G4 * Hh * 2);   // [N=1536][K=384]
  half_t* whhr16  = (half_t*)alloc((size_t)G4 * Hh * 2);
  half_t* s1_16   = (half_t*)alloc((size_t)Dd * Dd * 2);   // [N=768][K=768]
  half_t* rep16   = (half_t*)alloc((size_t)Dd * Dd * 2);
  half_t* w2p16   = (half_t*)alloc((size_t)Dd * KP * 2);   // [N=768][K=2400]
  half_t* w3p16   = (half_t*)alloc((size_t)Dd * KP * 2);
  float*  Gf      = (float*)alloc((size_t)BT * G4 * 4);
  float*  Gxr     = (float*)alloc((size_t)BT * G4 * 4);
  float*  gtmpF   = (float*)alloc((size_t)Bb * G4 * 4);
  float*  gtmpR   = (float*)alloc((size_t)Bb * G4 * 4);
  float*  cFR     = (float*)alloc((size_t)2 * Bb * Hh * 4);   // cF | cR
  half_t* h16FR   = (half_t*)alloc((size_t)2 * Bb * Hh * 2);  // h16F | h16R
  float*  hFh     = (float*)alloc((size_t)BT * Hh * 4);
  float*  hRh     = (float*)alloc((size_t)BT * Hh * 4);
  float*  enc     = (float*)alloc((size_t)BT * Dd * 4);
  half_t* enc16   = (half_t*)alloc((size_t)BT * Dd * 2);
  float*  Z       = (float*)alloc((size_t)BT * Dd * 4);
  float*  araw    = (float*)alloc((size_t)BT * 4);
  float*  alpha   = (float*)alloc((size_t)BT * 4);
  float*  U       = (float*)alloc((size_t)Bb * Dd * 4);
  int*    eidx    = (int*)alloc(NE * 4);
  int*    cidx    = (int*)alloc(NC * 4);
  float*  Ug      = (float*)alloc((size_t)(NE + NC) * Dd * 4);
  half_t* Ug16    = (half_t*)alloc((size_t)(NE + NC) * Dd * 2);
  float*  reptmp  = (float*)alloc((size_t)(NE + NC) * Dd * 4);
  float*  text    = (float*)alloc((size_t)(NE + NC) * Dd * 4);
  half_t* text16  = (half_t*)alloc((size_t)(NE + NC) * Dd * 2);
  float*  sc      = (float*)alloc((size_t)NE * NC * 4);
  float*  ce      = (float*)alloc((size_t)NE * NCH * Dd * 4);
  float*  nrm2    = (float*)alloc((size_t)NE * NCH * 4);
  half_t* delta2  = (half_t*)alloc((size_t)NE * NCH * KP * 2);
  float*  Z2      = (float*)alloc((size_t)NE * NCH * Dd * 4);
  float*  extra   = (float*)alloc((size_t)NE * NCH * 4);
  float*  nrm3    = (float*)alloc((size_t)NE * NC * 4);
  half_t* delta3  = (half_t*)alloc((size_t)NE * NC * KP * 2);
  float*  Z3      = (float*)alloc((size_t)NE * NC * Dd * 4);

  float*  cF = cFR;
  float*  cR = cFR + Bb * Hh;
  half_t* h16F = h16FR;
  half_t* h16R = h16FR + Bb * Hh;

  // ---- f16 conversions (weights already [N][K]; only w2/w3 need K-pad) ----
  k_f32_to_f16<<<CDIV(BT * Dd, 256), 256, 0, stream>>>(word, Xf16, BT * Dd);
  k_f32_to_f16<<<CDIV(G4 * Dd, 256), 256, 0, stream>>>(wihf, wihf16, G4 * Dd);
  k_f32_to_f16<<<CDIV(G4 * Dd, 256), 256, 0, stream>>>(wihr, wihr16, G4 * Dd);
  k_f32_to_f16<<<CDIV(G4 * Hh, 256), 256, 0, stream>>>(whhf, whhf16, G4 * Hh);
  k_f32_to_f16<<<CDIV(G4 * Hh, 256), 256, 0, stream>>>(whhr, whhr16, G4 * Hh);
  k_f32_to_f16<<<CDIV(Dd * Dd, 256), 256, 0, stream>>>(s1_w, s1_16, Dd * Dd);
  k_f32_to_f16<<<CDIV(Dd * Dd, 256), 256, 0, stream>>>(rep_w, rep16, Dd * Dd);
  k_pad_f16<<<CDIV(Dd * KP, 256), 256, 0, stream>>>(w2_w, w2p16, KD2, KP, Dd);
  k_pad_f16<<<CDIV(Dd * KP, 256), 256, 0, stream>>>(w3_w, w3p16, KD3, KP, Dd);
  k_zero_f32<<<CDIV(2 * Bb * Hh, 256), 256, 0, stream>>>(cFR, 2 * Bb * Hh);
  k_zero_f16<<<CDIV(2 * Bb * Hh, 256), 256, 0, stream>>>(h16FR, 2 * Bb * Hh);

  // ---- batched input projections (bias b_f folded into Gf) ----
  gemm(stream, Xf16, wihf16, bf, Gf, BT, G4, Dd, FLAG_BIAS);
  gemm(stream, Xf16, wihr16, nullptr, Gxr, BT, G4, Dd, 0);

  // ---- sequential BiLSTM recurrence (b_r folded into recurrent GEMM) ----
  for (int t = 0; t < Tt; ++t) {
    gemm(stream, h16F, whhf16, nullptr, gtmpF, Bb, G4, Hh, 0);
    gemm(stream, h16R, whhr16, br, gtmpR, Bb, G4, Hh, FLAG_BIAS);
    k_lstm_step<<<CDIV(2 * Bb * Hh, 256), 256, 0, stream>>>(
        gtmpF, gtmpR, Gf, Gxr, slen, t, cF, cR, hFh, hRh, h16F, h16R);
  }

  k_build_enc<<<CDIV(BT * Dd, 256), 256, 0, stream>>>(hFh, hRh, slen, enc, enc16);

  // ---- attention ----
  gemm(stream, enc16, s1_16, s1_b, Z, BT, Dd, Dd, FLAG_BIAS | FLAG_LEAKY);
  k_alpha_raw<<<BT / 8, 256, 0, stream>>>(Z, s2_w, slen, araw);
  k_alpha_norm<<<Bb, 64, 0, stream>>>(araw, alpha);
  k_weighted_sum<<<CDIV(Bb * Dd, 256), 256, 0, stream>>>(alpha, enc, U);

  // ---- gather + shared representation ----
  k_indices<<<1, 1, 0, stream>>>(bemo, bcau, eidx, cidx);
  k_gather_U<<<CDIV((NE + NC) * Dd, 256), 256, 0, stream>>>(U, eidx, cidx, Ug, Ug16);
  gemm(stream, Ug16, rep16, rep_b, reptmp, NE + NC, Dd, Dd, FLAG_BIAS | FLAG_LEAKY);
  k_select_text<<<CDIV((NE + NC) * Dd, 256), 256, 0, stream>>>(
      reptmp, Ug, share, text, text16);

  // ---- chunk attention: scores = emo @ cau^T  (cau rows ARE B[N][K]) ----
  gemm(stream, text16, text16 + (size_t)NE * Dd, nullptr, sc, NE, NC, Dd, 0);
  k_chunk_softmax<<<CDIV(NE * NCH, 256), 256, 0, stream>>>(sc);
  k_chunkemb<<<CDIV(NE * NCH * Dd, 256), 256, 0, stream>>>(sc, text, ce);

  // ---- phase 2 ----
  k_nrm2<<<NE * NCH / 8, 256, 0, stream>>>(text, ce, nrm2);
  k_delta2<<<CDIV(NE * NCH * KP, 256), 256, 0, stream>>>(text, ce, nrm2, pos, delta2);
  gemm(stream, delta2, w2p16, w2_b, Z2, NE * NCH, Dd, KP, FLAG_BIAS | FLAG_LEAKY);
  k_phase2<<<NE * NCH / 8, 256, 0, stream>>>(Z2, wo_w, wo_b, out_phase2, extra);

  // ---- phase 3 ----
  k_nrm3<<<NE * NC / 8, 256, 0, stream>>>(text, nrm3);
  k_delta3<<<CDIV(NE * NC * KP, 256), 256, 0, stream>>>(text, nrm3, dis, extra, delta3);
  gemm(stream, delta3, w3p16, w3_b, Z3, NE * NC, Dd, KP, FLAG_BIAS | FLAG_LEAKY);
  k_pcls<<<NE * NC / 8, 256, 0, stream>>>(Z3, cls_w, cls_b, out_p);

  // ---- labels ----
  k_labels<<<1, NE, 0, stream>>>(lab3, out_L);
}
